// FPModule_83296595738854
// MI455X (gfx1250) — compile-verified, matrix operand order checked by reading, not run on hardware
//
#include <hip/hip_runtime.h>
#include <hip/hip_bf16.h>
#include <math.h>
#include <stdint.h>

// ---------------------------------------------------------------------------
// PointNet++ FP-module for MI455X (gfx1250, wave32, WMMA + TDM).
//   knn_interpolate(k=3) -> concat -> [Linear 384->256 + ReLU + BN] ->
//   [Linear 256->256 + ReLU + BN]
// HBM-bound (~300 MB @ 23.3 TB/s ~ 15 us). GEMMs (21.5 GFLOP) run on
// v_wmma_f32_16x16x32_f16, f32 accumulate, f16-streamed activations.
// Wave = 64x16 output block: 4 independent accumulators share one B fragment
// (2.5 b128 loads per wmma instead of 8; no WMMA->WMMA RAW hazard NOPs).
// KNN stages 48 KB of coarse positions into LDS via the Tensor Data Mover.
// ---------------------------------------------------------------------------

typedef _Float16 v8h  __attribute__((ext_vector_type(8)));
typedef _Float16 v16h __attribute__((ext_vector_type(16)));
typedef float    v8f  __attribute__((ext_vector_type(8)));
typedef unsigned int v4u __attribute__((ext_vector_type(4)));
typedef int      v8i  __attribute__((ext_vector_type(8)));
typedef int      v4i  __attribute__((ext_vector_type(4)));

static constexpr int kB     = 4;       // batches
static constexpr int kNc    = 4096;    // coarse points / batch
static constexpr int kNf    = 16384;   // fine points / batch
static constexpr int kM     = 65536;   // total fine points (rows)
static constexpr int kCin   = 256;     // coarse feature channels
static constexpr int kCskip = 128;     // skip feature channels
static constexpr int kK1    = 384;     // GEMM1 K  (256 + 128)
static constexpr int kN     = 256;     // output channels both layers
static constexpr float kEps = 1e-5f;

#if defined(__has_builtin)
#if __has_builtin(__builtin_amdgcn_tensor_load_to_lds) && \
    __has_builtin(__builtin_amdgcn_s_wait_tensorcnt)
#define USE_TDM 1
#endif
#endif

// ---------------------------------------------------------------------------
// 1) KNN top-3. Coarse positions of one batch (12288 f32 = 48 KB) are staged
//    into LDS by the Tensor Data Mover (one 1D-tile descriptor, issued by
//    wave 0, completion via s_wait_tensorcnt). All lanes then scan the same
//    j => conflict-free LDS broadcast reads.
// ---------------------------------------------------------------------------
__global__ void knn_top3_kernel(const float* __restrict__ pos,
                                const float* __restrict__ pos_skip,
                                int* __restrict__ idx_out,
                                float* __restrict__ w_out) {
  __shared__ float ps[kNc * 3];
  const int b    = blockIdx.x >> 6;             // 64 blocks per batch
  const int fblk = (blockIdx.x & 63) * 256;
  const float* pb = pos + (size_t)b * kNc * 3;

#ifdef USE_TDM
  if (threadIdx.x < 32) {                       // one wave issues the DMA
    const uint64_t ga  = (uint64_t)(uintptr_t)pb;
    const uint32_t lds = (uint32_t)(uintptr_t)&ps[0];  // low 32b = LDS offset
    // D# group0: count=1 | lds_addr | global_addr[56:0] | type=2 (bits 127:126)
    const v4u g0 = { 1u, lds, (uint32_t)ga,
                     (uint32_t)(((ga >> 32) & 0x1FFFFFFu) | (2u << 30)) };
    // D# group1: data_size=4B (code 2); tensor_dim0 = tile_dim0 = 12288;
    // tensor_dim1 = 1; strides = 12288; no pad / iterate / multicast.
    const v8i g1 = { 0x00020000, 0x30000000, 0x00010000, 0x30000000,
                     0, 0x00003000, 0x30000000, 0 };
    const v4i gz = { 0, 0, 0, 0 };
#if __clang_major__ >= 23
    const v8i gz8 = { 0, 0, 0, 0, 0, 0, 0, 0 };
    __builtin_amdgcn_tensor_load_to_lds(g0, g1, gz, gz, gz8, 0);
#else
    __builtin_amdgcn_tensor_load_to_lds(g0, g1, gz, gz, 0);
#endif
    __builtin_amdgcn_s_wait_tensorcnt(0);       // TENSORcnt == 0 -> LDS valid
  }
#else
  for (int i = threadIdx.x; i < kNc * 3; i += 256) ps[i] = pb[i];
#endif
  __syncthreads();

  const int f = b * kNf + fblk + threadIdx.x;
  const float qx = pos_skip[(size_t)f * 3 + 0];
  const float qy = pos_skip[(size_t)f * 3 + 1];
  const float qz = pos_skip[(size_t)f * 3 + 2];

  float d0 = 3.4e38f, d1 = 3.4e38f, d2 = 3.4e38f;
  int   i0 = 0, i1 = 0, i2 = 0;
  for (int j = 0; j < kNc; ++j) {
    const float dx = qx - ps[j * 3 + 0];
    const float dy = qy - ps[j * 3 + 1];
    const float dz = qz - ps[j * 3 + 2];
    const float d  = dx * dx + dy * dy + dz * dz;
    if (d < d2) {
      if (d < d1) {
        d2 = d1; i2 = i1;
        if (d < d0) { d1 = d0; i1 = i0; d0 = d; i0 = j; }
        else        { d1 = d;  i1 = j; }
      } else { d2 = d; i2 = j; }
    }
  }
  const float w0 = 1.f / fmaxf(d0, 1e-16f);
  const float w1 = 1.f / fmaxf(d1, 1e-16f);
  const float w2 = 1.f / fmaxf(d2, 1e-16f);
  const float inv = 1.f / (w0 + w1 + w2);
  idx_out[(size_t)f * 3 + 0] = b * kNc + i0;
  idx_out[(size_t)f * 3 + 1] = b * kNc + i1;
  idx_out[(size_t)f * 3 + 2] = b * kNc + i2;
  w_out[(size_t)f * 3 + 0] = w0 * inv;
  w_out[(size_t)f * 3 + 1] = w1 * inv;
  w_out[(size_t)f * 3 + 2] = w2 * inv;
}

// ---------------------------------------------------------------------------
// 2) Build A1[f, 0:384] = f16( [interp(x), x_skip] ).  x is 16 MB -> L2-hot.
// ---------------------------------------------------------------------------
__global__ void build_a1_kernel(const float* __restrict__ x,
                                const float* __restrict__ x_skip,
                                const int* __restrict__ idx,
                                const float* __restrict__ w,
                                _Float16* __restrict__ A1) {
  const size_t gid = (size_t)blockIdx.x * 256 + threadIdx.x; // kM*kK1 threads
  const int f = (int)(gid / kK1);
  const int c = (int)(gid % kK1);
  float v;
  if (c < kCin) {
    const int ia = idx[(size_t)f * 3 + 0];
    const int ib = idx[(size_t)f * 3 + 1];
    const int ic = idx[(size_t)f * 3 + 2];
    v = w[(size_t)f * 3 + 0] * x[(size_t)ia * kCin + c]
      + w[(size_t)f * 3 + 1] * x[(size_t)ib * kCin + c]
      + w[(size_t)f * 3 + 2] * x[(size_t)ic * kCin + c];
  } else {
    v = x_skip[(size_t)f * kCskip + (c - kCin)];
  }
  A1[gid] = (_Float16)v;
}

// ---------------------------------------------------------------------------
// 3) Weight prep: Wt[n*K + k] = f16(W[k*N + n])  (B consumed column-major).
// ---------------------------------------------------------------------------
__global__ void prep_weights_kernel(const float* __restrict__ W1,
                                    const float* __restrict__ W2,
                                    _Float16* __restrict__ Wt1,
                                    _Float16* __restrict__ Wt2) {
  const int gid = blockIdx.x * 256 + threadIdx.x;
  if (gid < kK1 * kN) {
    const int k = gid / kN, n = gid % kN;
    Wt1[(size_t)n * kK1 + k] = (_Float16)W1[gid];
  } else if (gid < kK1 * kN + kN * kN) {
    const int g = gid - kK1 * kN;
    const int k = g / kN, n = g % kN;
    Wt2[(size_t)n * kN + k] = (_Float16)W2[g];
  }
}

// ---------------------------------------------------------------------------
// 4) WMMA GEMM: out[M,256] = relu(A[M,K] @ W[K,256] + bias), f32 out.
//    Wave = 64(M) x 16(N): 4 independent v8f accumulators share one B
//    fragment per 32-wide k-step. Fragment addressing per ISA 7.12.2
//    (16-bit A 16x32): lane L holds row L&15; lanes<16 carry K
//    {kt..kt+7, kt+16..kt+23}, lanes>=16 carry {kt+8..kt+15, kt+24..kt+31}
//    -> two 16B loads per fragment. B identical against Wt[N,K].
//    C/D: VGPR r -> row base + r + 8*(lane>=16), col tn + (lane&15).
// ---------------------------------------------------------------------------
__device__ __forceinline__ v16h load_frag(const _Float16* p) {
  const v8h lo = *(const v8h*)(p);
  const v8h hi = *(const v8h*)(p + 16);
  v16h r;
#pragma unroll
  for (int i = 0; i < 8; ++i) { r[i] = lo[i]; r[8 + i] = hi[i]; }
  return r;
}

__global__ void gemm_bias_relu_kernel(const _Float16* __restrict__ A,
                                      const _Float16* __restrict__ Wt,
                                      const float* __restrict__ bias,
                                      float* __restrict__ out,
                                      int Kdim) {
  const int wave     = blockIdx.x * 8 + (threadIdx.x >> 5);
  const int lane     = threadIdx.x & 31;
  const int tn       = (wave & 15) << 4;   // 256/16 = 16 tiles in N
  const int tm       = (wave >> 4) << 6;   // 64 rows per wave
  const int half_sel = lane >> 4;
  const int r        = lane & 15;

  const _Float16* ap = A  + (size_t)(tm + r) * Kdim + 8 * half_sel;
  const _Float16* bp = Wt + (size_t)(tn + r) * Kdim + 8 * half_sel;
  const size_t rstep = (size_t)16 * Kdim;  // 16 rows of A

  v8f acc[4];
#pragma unroll
  for (int t = 0; t < 4; ++t) acc[t] = (v8f){};

  for (int kt = 0; kt < Kdim; kt += 32) {
    const v16h bv = load_frag(bp + kt);
    __builtin_prefetch(ap + kt + 64, 0, 1);               // global_prefetch_b8
    __builtin_prefetch(ap + 2 * rstep + kt + 64, 0, 1);
#pragma unroll
    for (int t = 0; t < 4; ++t) {
      const v16h av = load_frag(ap + (size_t)t * rstep + kt);
      acc[t] = __builtin_amdgcn_wmma_f32_16x16x32_f16(
          false, av, false, bv, (short)0, acc[t], false, false);
    }
  }

  const int col = tn + r;
  const float bb = bias[col];
#pragma unroll
  for (int t = 0; t < 4; ++t) {
    const int rowbase = tm + t * 16 + 8 * half_sel;
#pragma unroll
    for (int rr = 0; rr < 8; ++rr) {
      out[(size_t)(rowbase + rr) * kN + col] = fmaxf(acc[t][rr] + bb, 0.f);
    }
  }
}

// ---------------------------------------------------------------------------
// 5) Per-channel sum / sum^2 (training BN). thread = channel -> coalesced.
// ---------------------------------------------------------------------------
__global__ void col_stats_kernel(const float* __restrict__ v,
                                 float* __restrict__ sum,
                                 float* __restrict__ ssq) {
  const int t = threadIdx.x;                    // channel 0..255
  const size_t row0 = (size_t)blockIdx.x * 256; // 256 rows per block
  float s = 0.f, q = 0.f;
  for (int rr = 0; rr < 256; ++rr) {
    const float x = v[(row0 + rr) * kN + t];
    s += x;
    q += x * x;
  }
  atomicAdd(&sum[t], s);
  atomicAdd(&ssq[t], q);
}

// ---------------------------------------------------------------------------
// 6) BN apply. Variant a: f16 A-matrix for next GEMM. Variant b: f32 in-place
//    on d_out (d_out fully rewritten by GEMM2 each call -> replay-safe).
// ---------------------------------------------------------------------------
__global__ void bn_to_f16_kernel(const float* __restrict__ v,
                                 const float* __restrict__ sum,
                                 const float* __restrict__ ssq,
                                 const float* __restrict__ g,
                                 const float* __restrict__ beta,
                                 _Float16* __restrict__ out) {
  const size_t e = (size_t)blockIdx.x * 256 + threadIdx.x;
  const int ch = (int)(e & 255);
  const float mean = sum[ch] * (1.f / kM);
  const float var  = ssq[ch] * (1.f / kM) - mean * mean;
  const float y = g[ch] * (v[e] - mean) * rsqrtf(var + kEps) + beta[ch];
  out[e] = (_Float16)y;
}

__global__ void bn_inplace_kernel(float* __restrict__ v,
                                  const float* __restrict__ sum,
                                  const float* __restrict__ ssq,
                                  const float* __restrict__ g,
                                  const float* __restrict__ beta) {
  const size_t e = (size_t)blockIdx.x * 256 + threadIdx.x;
  const int ch = (int)(e & 255);
  const float mean = sum[ch] * (1.f / kM);
  const float var  = ssq[ch] * (1.f / kM) - mean * mean;
  v[e] = g[ch] * (v[e] - mean) * rsqrtf(var + kEps) + beta[ch];
}

__global__ void zero_stats_kernel(float* __restrict__ p) {
  p[blockIdx.x * 256 + threadIdx.x] = 0.f;      // grid 4 -> 1024 floats
}

// ---------------------------------------------------------------------------
// Launch. Inputs (all f32, setup_inputs order):
//   0 x[16384*256] 1 pos[16384*3] 2 x_skip[65536*128] 3 pos_skip[65536*3]
//   4 W1[384*256] 5 b1 6 g1 7 beta1 8 W2[256*256] 9 b2 10 g2 11 beta2
// ---------------------------------------------------------------------------
extern "C" void kernel_launch(void* const* d_in, const int* in_sizes, int n_in,
                              void* d_out, int out_size, void* d_ws, size_t ws_size,
                              hipStream_t stream) {
  const float* x        = (const float*)d_in[0];
  const float* pos      = (const float*)d_in[1];
  const float* x_skip   = (const float*)d_in[2];
  const float* pos_skip = (const float*)d_in[3];
  const float* W1       = (const float*)d_in[4];
  const float* b1       = (const float*)d_in[5];
  const float* g1       = (const float*)d_in[6];
  const float* beta1    = (const float*)d_in[7];
  const float* W2       = (const float*)d_in[8];
  const float* b2       = (const float*)d_in[9];
  const float* g2       = (const float*)d_in[10];
  const float* beta2    = (const float*)d_in[11];
  float* out = (float*)d_out;

  // Workspace layout (bytes)
  char* ws = (char*)d_ws;
  float*    sum1 = (float*)(ws + 0);
  float*    ssq1 = (float*)(ws + 1024);
  float*    sum2 = (float*)(ws + 2048);
  float*    ssq2 = (float*)(ws + 3072);
  int*      idx3 = (int*)  (ws + 4096);                        // 65536*3 int
  float*    wgt3 = (float*)(ws + 4096 + 786432);               // 65536*3 f32
  _Float16* A1   = (_Float16*)(ws + 1576960);                  // 65536*384 f16 (48MB)
  _Float16* A2   = A1;                                         // reuse after GEMM1
  _Float16* Wt1  = (_Float16*)(ws + 1576960 + 50331648);       // 384*256 f16
  _Float16* Wt2  = (_Float16*)(ws + 1576960 + 50331648 + 196608);
  float*    r1   = (float*)(ws + 1576960 + 50331648 + 196608 + 131072); // 65536*256 f32

  zero_stats_kernel<<<4, 256, 0, stream>>>((float*)ws);

  knn_top3_kernel<<<256, 256, 0, stream>>>(pos, pos_skip, idx3, wgt3);

  prep_weights_kernel<<<(kK1 * kN + kN * kN + 255) / 256, 256, 0, stream>>>(
      W1, W2, Wt1, Wt2);

  build_a1_kernel<<<(kM * kK1) / 256, 256, 0, stream>>>(x, x_skip, idx3, wgt3, A1);

  // Layer 1: 65536x384 @ 384x256 -> relu -> r1
  gemm_bias_relu_kernel<<<(kM / 64) * (kN / 16) / 8, 256, 0, stream>>>(
      A1, Wt1, b1, r1, kK1);
  col_stats_kernel<<<kM / 256, 256, 0, stream>>>(r1, sum1, ssq1);
  bn_to_f16_kernel<<<(kM * kN) / 256, 256, 0, stream>>>(r1, sum1, ssq1, g1, beta1, A2);

  // Layer 2: 65536x256 @ 256x256 -> relu -> d_out, then BN in place
  gemm_bias_relu_kernel<<<(kM / 64) * (kN / 16) / 8, 256, 0, stream>>>(
      A2, Wt2, b2, out, kN);
  col_stats_kernel<<<kM / 256, 256, 0, stream>>>(out, sum2, ssq2);
  bn_inplace_kernel<<<(kM * kN) / 256, 256, 0, stream>>>(out, sum2, ssq2, g2, beta2);
}
